// NeuralODE_58480274702439
// MI455X (gfx1250) — compile-verified, matrix operand order checked by reading, not run on hardware
//
#include <hip/hip_runtime.h>
#include <hip/hip_bf16.h>

typedef __bf16 bf16_t;
typedef __attribute__((ext_vector_type(16))) __bf16 v16bf;
typedef __attribute__((ext_vector_type(8)))  float  v8f;
typedef __attribute__((ext_vector_type(4)))  int    v4i;

union Frag16 {
    v16bf bf;
    v4i   q[2];
};

__device__ __forceinline__ float fast_tanh(float x) {
#if __has_builtin(__builtin_amdgcn_tanhf)
    return __builtin_amdgcn_tanhf(x);   // V_TANH_F32 (gfx1250 transcendental)
#else
    return tanhf(x);
#endif
}

// ---------------------------------------------------------------------------
// Pack a row-major f32 weight matrix [K x N] (row stride = N) into bf16 WMMA
// B-fragment layout. Fragment (nt, kt) covers cols nt*16..+15, rows kt*32..+31.
// Within a fragment, lane L holds column n = nt*16 + (L&15); its 16 bf16
// values are K = kb..kb+7 then kb+16..kb+23 where kb = kt*32 + (L>>4)*8.
// Stored flat: Wp[ ((nt*KT + kt)*32 + L)*16 + j ], so a consuming lane does
// two contiguous b128 loads.
// ---------------------------------------------------------------------------
__global__ void node_pack_weight(const float* __restrict__ W, bf16_t* __restrict__ Wp,
                                 int K, int N)
{
    int tid = blockIdx.x * blockDim.x + threadIdx.x;
    int KT  = K >> 5;
    int total = (N >> 4) * KT * 32;
    if (tid >= total) return;
    int lane = tid & 31;
    int frag = tid >> 5;          // nt*KT + kt
    int kt   = frag % KT;
    int nt   = frag / KT;
    int n    = (nt << 4) + (lane & 15);
    int kb   = (kt << 5) + ((lane >> 4) << 3);
    bf16_t* dst = Wp + (size_t)tid * 16;
#pragma unroll
    for (int j = 0; j < 16; ++j) {
        int k = (j < 8) ? (kb + j) : (kb + 8 + j);   // kb..kb+7, kb+16..kb+23
        dst[j] = (bf16_t)W[(size_t)k * N + n];
    }
}

__global__ void node_init_z(const float* __restrict__ z0, float* __restrict__ zf,
                            bf16_t* __restrict__ zb, int n)
{
    int i = blockIdx.x * blockDim.x + threadIdx.x;
    if (i < n) {
        float v = z0[i];
        zf[i] = v;
        zb[i] = (bf16_t)v;
    }
}

// ---------------------------------------------------------------------------
// GEMM1: H = tanh(z * W1 + (b1 + t*W1_last)).  M=1024, K=1024, N=2048.
// 8 waves/block; each wave computes a 32(M) x 64(N) tile (2x4 accumulators).
// Accumulators are initialized with the per-column bias, so the epilogue is
// just tanh + bf16 store.
// ---------------------------------------------------------------------------
__global__ void __launch_bounds__(256)
node_gemm1_tanh(const bf16_t* __restrict__ A,      // z bf16 [1024 x 1024]
                const bf16_t* __restrict__ Bp,     // packed W1 bf16
                const float*  __restrict__ b1,     // [2048]
                const float*  __restrict__ W1last, // [2048] (row 1024 of W1)
                bf16_t*       __restrict__ H,      // [1024 x 2048] bf16
                float t)
{
    const int K = 1024, N = 2048;
    const int KT = K >> 5;                 // 32
    int lane  = threadIdx.x & 31;
    int gw    = blockIdx.x * 8 + (threadIdx.x >> 5);
    int mTile = gw & 31;                   // 32 m-tiles of 32 rows
    int nTile = gw >> 5;                   // 32 n-tiles of 64 cols
    int mBase = mTile << 5;
    int nBase = nTile << 6;
    int colLane = lane & 15;

    // Init accumulators with the folded bias (b1 + t * W1_last), per column.
    v8f acc[2][4];
#pragma unroll
    for (int j = 0; j < 4; ++j) {
        int n = nBase + (j << 4) + colLane;
        float bias = b1[n] + t * W1last[n];
#pragma unroll
        for (int i = 0; i < 2; ++i)
#pragma unroll
            for (int r = 0; r < 8; ++r)
                acc[i][j][r] = bias;
    }

    int mRow = mBase + colLane;
    int kSub = (lane >> 4) << 3;
    const bf16_t* aRow0 = A + (size_t)mRow * K;
    const bf16_t* aRow1 = A + (size_t)(mRow + 16) * K;
    int nt16 = nBase >> 4;

    for (int kt = 0; kt < KT; ++kt) {
        int kb = (kt << 5) + kSub;
        Frag16 a0, a1;
        a0.q[0] = *(const v4i*)(aRow0 + kb);
        a0.q[1] = *(const v4i*)(aRow0 + kb + 16);
        a1.q[0] = *(const v4i*)(aRow1 + kb);
        a1.q[1] = *(const v4i*)(aRow1 + kb + 16);
        if (kt + 1 < KT)
            __builtin_prefetch(Bp + ((size_t)(nt16 * KT + kt + 1) * 32 + lane) * 16, 0, 3);
#pragma unroll
        for (int j = 0; j < 4; ++j) {
            const bf16_t* bp = Bp + ((size_t)((nt16 + j) * KT + kt) * 32 + lane) * 16;
            Frag16 b;
            b.q[0] = *(const v4i*)bp;
            b.q[1] = *(const v4i*)(bp + 8);
            acc[0][j] = __builtin_amdgcn_wmma_f32_16x16x32_bf16(
                false, a0.bf, false, b.bf, (short)0, acc[0][j], false, false);
            acc[1][j] = __builtin_amdgcn_wmma_f32_16x16x32_bf16(
                false, a1.bf, false, b.bf, (short)0, acc[1][j], false, false);
        }
    }

    // Epilogue: tanh, store bf16.
    int rowOff = (lane >> 4) << 3;
#pragma unroll
    for (int j = 0; j < 4; ++j) {
        int n = nBase + (j << 4) + colLane;
#pragma unroll
        for (int i = 0; i < 2; ++i) {
#pragma unroll
            for (int r = 0; r < 8; ++r) {
                int row = mBase + (i << 4) + rowOff + r;
                H[(size_t)row * N + n] = (bf16_t)fast_tanh(acc[i][j][r]);
            }
        }
    }
}

// ---------------------------------------------------------------------------
// GEMM2 + Euler update: z_out = z_in + h * (H * W2 + b2).  M=1024,K=2048,N=1024
// ---------------------------------------------------------------------------
__global__ void __launch_bounds__(256)
node_gemm2_euler(const bf16_t* __restrict__ A,     // H bf16 [1024 x 2048]
                 const bf16_t* __restrict__ Bp,    // packed W2 bf16
                 const float*  __restrict__ b2,    // [1024]
                 const float*  __restrict__ zin,   // [1024 x 1024] f32
                 float*        __restrict__ zoutF, // [1024 x 1024] f32
                 bf16_t*       __restrict__ zoutB, // [1024 x 1024] bf16
                 float h)
{
    const int K = 2048, N = 1024;
    const int KT = K >> 5;                 // 64
    int lane  = threadIdx.x & 31;
    int gw    = blockIdx.x * 8 + (threadIdx.x >> 5);
    int mTile = gw & 31;                   // 32 m-tiles
    int nTile = gw >> 5;                   // 16 n-tiles of 64 cols
    int mBase = mTile << 5;
    int nBase = nTile << 6;
    int colLane = lane & 15;

    // Init accumulators with b2 (per column).
    v8f acc[2][4];
#pragma unroll
    for (int j = 0; j < 4; ++j) {
        float bias = b2[nBase + (j << 4) + colLane];
#pragma unroll
        for (int i = 0; i < 2; ++i)
#pragma unroll
            for (int r = 0; r < 8; ++r)
                acc[i][j][r] = bias;
    }

    int mRow = mBase + colLane;
    int kSub = (lane >> 4) << 3;
    const bf16_t* aRow0 = A + (size_t)mRow * K;
    const bf16_t* aRow1 = A + (size_t)(mRow + 16) * K;
    int nt16 = nBase >> 4;

    for (int kt = 0; kt < KT; ++kt) {
        int kb = (kt << 5) + kSub;
        Frag16 a0, a1;
        a0.q[0] = *(const v4i*)(aRow0 + kb);
        a0.q[1] = *(const v4i*)(aRow0 + kb + 16);
        a1.q[0] = *(const v4i*)(aRow1 + kb);
        a1.q[1] = *(const v4i*)(aRow1 + kb + 16);
        if (kt + 1 < KT)
            __builtin_prefetch(Bp + ((size_t)(nt16 * KT + kt + 1) * 32 + lane) * 16, 0, 3);
#pragma unroll
        for (int j = 0; j < 4; ++j) {
            const bf16_t* bp = Bp + ((size_t)((nt16 + j) * KT + kt) * 32 + lane) * 16;
            Frag16 b;
            b.q[0] = *(const v4i*)bp;
            b.q[1] = *(const v4i*)(bp + 8);
            acc[0][j] = __builtin_amdgcn_wmma_f32_16x16x32_bf16(
                false, a0.bf, false, b.bf, (short)0, acc[0][j], false, false);
            acc[1][j] = __builtin_amdgcn_wmma_f32_16x16x32_bf16(
                false, a1.bf, false, b.bf, (short)0, acc[1][j], false, false);
        }
    }

    int rowOff = (lane >> 4) << 3;
#pragma unroll
    for (int j = 0; j < 4; ++j) {
        int n = nBase + (j << 4) + colLane;
#pragma unroll
        for (int i = 0; i < 2; ++i) {
#pragma unroll
            for (int r = 0; r < 8; ++r) {
                int row = mBase + (i << 4) + rowOff + r;
                size_t idx = (size_t)row * N + n;
                float znew = zin[idx] + h * acc[i][j][r];
                zoutF[idx] = znew;
                zoutB[idx] = (bf16_t)znew;
            }
        }
    }
}

extern "C" void kernel_launch(void* const* d_in, const int* in_sizes, int n_in,
                              void* d_out, int out_size, void* d_ws, size_t ws_size,
                              hipStream_t stream)
{
    const float* z0 = (const float*)d_in[0];
    // d_in[1] = t (linspace(0,1,5)); schedule reproduced on host
    const float* W1 = (const float*)d_in[2];   // [1025 x 2048]
    const float* b1 = (const float*)d_in[3];   // [2048]
    const float* W2 = (const float*)d_in[4];   // [2048 x 1024]
    const float* b2 = (const float*)d_in[5];   // [1024]
    float* out = (float*)d_out;                // [1024 x 1024]

    char* ws = (char*)d_ws;
    bf16_t* W1p = (bf16_t*)(ws);                       // 4 MiB
    bf16_t* W2p = (bf16_t*)(ws + (4u  << 20));         // 4 MiB
    float*  zf  = (float*) (ws + (8u  << 20));         // 4 MiB
    bf16_t* zb  = (bf16_t*)(ws + (12u << 20));         // 2 MiB
    bf16_t* Hb  = (bf16_t*)(ws + (14u << 20));         // 4 MiB

    // Pack weights (bf16, WMMA B-fragment layout). 131072 lanes each.
    node_pack_weight<<<512, 256, 0, stream>>>(W1, W1p, 1024, 2048);
    node_pack_weight<<<512, 256, 0, stream>>>(W2, W2p, 2048, 1024);
    node_init_z<<<(1024 * 1024) / 256, 256, 0, stream>>>(z0, zf, zb, 1024 * 1024);

    const float hstep = 0.25f / 5.0f;   // h_max-derived Euler step, 20 steps total
    for (int seg = 0; seg < 4; ++seg) {
        float t = 0.25f * (float)seg;
        for (int s = 0; s < 5; ++s) {
            node_gemm1_tanh<<<128, 256, 0, stream>>>(zb, W1p, b1,
                                                     W1 + (size_t)1024 * 2048, Hb, t);
            bool last = (seg == 3 && s == 4);
            node_gemm2_euler<<<64, 256, 0, stream>>>(Hb, W2p, b2, zf,
                                                     last ? out : zf, zb, hstep);
            t += hstep;
        }
    }
}
